// EvoformerBlockSS_72627896975606
// MI455X (gfx1250) — compile-verified, matrix operand order checked by reading, not run on hardware
//
#include <hip/hip_runtime.h>
#include <math.h>

#define LQ 192
#define WPB 4          // waves per block (wave32)
#define ZROWS 36864    // 192*192

typedef __attribute__((ext_vector_type(16))) _Float16 v16h;
typedef __attribute__((ext_vector_type(8)))  float    v8f;
typedef __attribute__((ext_vector_type(8)))  _Float16 h8;

enum { EPI_STORE=0, EPI_ADD=1, EPI_RELU=2, EPI_SIGMOID=3, EPI_SIG_MUL=4, EPI_GATE_MUL_ADD=5 };

__device__ inline v8f wmma_f16(v16h a, v16h b, v8f c) {
  // (neg_a, A, neg_b, B, c_mod, C, reuse_a, reuse_b) -> v_wmma_f32_16x16x32_f16
  return __builtin_amdgcn_wmma_f32_16x16x32_f16(false, a, false, b, (short)0, c, false, false);
}

__device__ inline float dsigmoid(float x) { return 1.0f / (1.0f + __expf(-x)); }
__device__ inline int   permRow(int m)    { return (m % LQ) * LQ + (m / LQ); }

// ---- fragment loaders --------------------------------------------------
// A fragment: 16x32 tile; element (m,k) at base[m*ms + k*ks]
// lanes 0-15: M=lane, K={0..7,16..23}; lanes 16-31: M=lane-16, K={8..15,24..31}
// K1 variant: ks==1, 16B-aligned rows -> float4 (b128) loads.
template <bool K1>
__device__ inline v16h load_frag_a(const float* base, int ms, int ks, int lane) {
  const float* rp = base + (lane & 15) * ms;
  int kh = (lane >> 4) ? 8 : 0;
  v16h f;
  if (K1) {
#pragma unroll
    for (int g = 0; g < 2; ++g) {
      const float4* q4 = (const float4*)(rp + kh + g * 16);
      float4 a = q4[0], b = q4[1];
      f[g * 8 + 0] = (_Float16)a.x; f[g * 8 + 1] = (_Float16)a.y;
      f[g * 8 + 2] = (_Float16)a.z; f[g * 8 + 3] = (_Float16)a.w;
      f[g * 8 + 4] = (_Float16)b.x; f[g * 8 + 5] = (_Float16)b.y;
      f[g * 8 + 6] = (_Float16)b.z; f[g * 8 + 7] = (_Float16)b.w;
    }
  } else {
#pragma unroll
    for (int g = 0; g < 2; ++g)
#pragma unroll
      for (int t = 0; t < 8; ++t)
        f[g * 8 + t] = (_Float16)rp[(kh + g * 16 + t) * ks];
  }
  return f;
}

// B fragment: 32x16 (KxN) tile; element (n,k) at base[n*ns + k*ks]; zeros if ncol >= nvalid
template <bool K1>
__device__ inline v16h load_frag_b(const float* base, int ns, int ks, int lane,
                                   int ncol, int nvalid) {
  v16h f;
  if (ncol >= nvalid) {
#pragma unroll
    for (int i = 0; i < 16; ++i) f[i] = (_Float16)0.0f;
    return f;
  }
  const float* rp = base + (lane & 15) * ns;
  int kh = (lane >> 4) ? 8 : 0;
  if (K1) {
#pragma unroll
    for (int g = 0; g < 2; ++g) {
      const float4* q4 = (const float4*)(rp + kh + g * 16);
      float4 a = q4[0], b = q4[1];
      f[g * 8 + 0] = (_Float16)a.x; f[g * 8 + 1] = (_Float16)a.y;
      f[g * 8 + 2] = (_Float16)a.z; f[g * 8 + 3] = (_Float16)a.w;
      f[g * 8 + 4] = (_Float16)b.x; f[g * 8 + 5] = (_Float16)b.y;
      f[g * 8 + 6] = (_Float16)b.z; f[g * 8 + 7] = (_Float16)b.w;
    }
  } else {
#pragma unroll
    for (int g = 0; g < 2; ++g)
#pragma unroll
      for (int t = 0; t < 8; ++t)
        f[g * 8 + t] = (_Float16)rp[(kh + g * 16 + t) * ks];
  }
  return f;
}

// ---- generic strided WMMA GEMM ----------------------------------------
struct GemmP {
  const float* A; int Ab, Am, Ak;
  const float* B; int Bb, Bn, Bk;
  float*       C; int Cb, Cm, Cn;
  const float* bias;                 // per-output-column bias (or null)
  const float* other; int Om, On;    // second operand for fused epilogues
  int M, N, K;
  float alpha;
  int epi;
  int maskN;   // 0 = none; else only columns < maskN are valid/stored
  int permC;   // 1 = permute output row (i*192+j -> j*192+i)
};

template <bool AK1, bool BK1>
__global__ void gemm_kernel(GemmP p) {
  int lane = threadIdx.x & 31;
  int wave = __builtin_amdgcn_readfirstlane(threadIdx.x >> 5);  // wave-uniform -> SGPR
  int mt_n = p.M >> 4, nt_n = p.N >> 4;
  int tile = blockIdx.x * WPB + wave;
  if (tile >= mt_n * nt_n) return;
  int mt = tile / nt_n, nt = tile % nt_n;
  int b = blockIdx.y;
  const float* Abase = p.A + b * p.Ab + mt * 16 * p.Am;
  const float* Bbase = p.B + b * p.Bb + nt * 16 * p.Bn;
  int nvalid = p.maskN ? p.maskN : p.N;
  int ncol = nt * 16 + (lane & 15);
  v8f acc = {0.f, 0.f, 0.f, 0.f, 0.f, 0.f, 0.f, 0.f};
  for (int k = 0; k < p.K; k += 32) {
    if (k + 32 < p.K)  // prefetch next A K-slab (global_prefetch_b8)
      __builtin_prefetch(Abase + (k + 32) * p.Ak, 0, 3);
    v16h af = load_frag_a<AK1>(Abase + k * p.Ak, p.Am, p.Ak, lane);
    v16h bf = load_frag_b<BK1>(Bbase + k * p.Bk, p.Bn, p.Bk, lane, ncol, nvalid);
    acc = wmma_f16(af, bf, acc);
  }
  if (p.maskN && ncol >= p.maskN) return;
  int half = lane >> 4;
  float bn = p.bias ? p.bias[ncol] : 0.f;
#pragma unroll
  for (int r = 0; r < 8; ++r) {
    int m = mt * 16 + r + half * 8;
    float val = acc[r] * p.alpha + bn;
    int mrow = p.permC ? permRow(m) : m;
    float* cp = p.C + b * p.Cb + mrow * p.Cm + ncol * p.Cn;
    switch (p.epi) {
      case EPI_STORE:   *cp = val; break;
      case EPI_ADD:     *cp += val; break;
      case EPI_RELU:    *cp = fmaxf(val, 0.f); break;
      case EPI_SIGMOID: *cp = dsigmoid(val); break;
      case EPI_SIG_MUL: *cp = dsigmoid(val) * p.other[m * p.Om + ncol * p.On]; break;
      case EPI_GATE_MUL_ADD:
        *cp += p.other[m * p.Om + ncol * p.On] * val; break;
    }
  }
}

// ---- LayerNorm ---------------------------------------------------------
__global__ void ln_kernel(const float* __restrict__ in, float* __restrict__ out,
                          const float* __restrict__ gam, const float* __restrict__ bet,
                          int C, int perm) {
  int row = blockIdx.x;
  int src = perm ? permRow(row) : row;
  const float* x = in + src * C;
  float* y = out + row * C;
  __shared__ float red[128];
  int t = threadIdx.x;
  float s = 0.f;
  for (int c = t; c < C; c += 128) s += x[c];
  red[t] = s; __syncthreads();
  for (int o = 64; o > 0; o >>= 1) { if (t < o) red[t] += red[t + o]; __syncthreads(); }
  float mu = red[0] / (float)C; __syncthreads();
  float v = 0.f;
  for (int c = t; c < C; c += 128) { float d = x[c] - mu; v += d * d; }
  red[t] = v; __syncthreads();
  for (int o = 64; o > 0; o >>= 1) { if (t < o) red[t] += red[t + o]; __syncthreads(); }
  float rs = rsqrtf(red[0] / (float)C + 1e-5f);
  for (int c = t; c < C; c += 128) y[c] = (x[c] - mu) * rs * gam[c] + bet[c];
}

// ---- fused gated attention --------------------------------------------
// per wave = 16 query rows of one (row-batch r, head h):
// QK via WMMA -> +bias,*scale -> wave32 softmax (shfl halves) -> probs to LDS(f16)
// -> AV via WMMA (probs re-read as b128 LDS vectors) -> out *= sigmoid(gate)
// __launch_bounds__(128,1): 12 v8f logits accumulators must stay in VGPRs (no spill)
__global__ void __launch_bounds__(WPB * 32, 1)
attn_kernel(const float* __restrict__ q, const float* __restrict__ k,
            const float* __restrict__ v, const float* __restrict__ gate,
            float* __restrict__ o, const float* __restrict__ bias,
            int H, int ld, int nwork, float scale) {
  __shared__ _Float16 pbuf[WPB][16][200];
  int lane = threadIdx.x & 31;
  int wave = __builtin_amdgcn_readfirstlane(threadIdx.x >> 5);
  int gid = blockIdx.x * WPB + wave;
  if (gid >= nwork) return;
  int it = gid % 12;
  int hh = (gid / 12) % H;
  int r = gid / (12 * H);
  int rowbase = r * LQ;
  int half = lane >> 4;
  int l15 = lane & 15;

  const float* qb = q + (rowbase + it * 16) * ld + hh * 32;
  const float* kb = k + rowbase * ld + hh * 32;
  v16h aq = load_frag_a<true>(qb, ld, 1, lane);
  v8f z8 = {0.f, 0.f, 0.f, 0.f, 0.f, 0.f, 0.f, 0.f};
  v8f lg[12];
#pragma unroll
  for (int nt = 0; nt < 12; ++nt) {
    v16h bk = load_frag_b<true>(kb + nt * 16 * ld, ld, 1, lane, 0, 16);
    lg[nt] = wmma_f16(aq, bk, z8);
  }
  const float* bb = bias + hh * ZROWS + (it * 16) * LQ;
#pragma unroll
  for (int nt = 0; nt < 12; ++nt)
#pragma unroll
    for (int r8 = 0; r8 < 8; ++r8)
      lg[nt][r8] = lg[nt][r8] * scale + bb[(r8 + half * 8) * LQ + nt * 16 + l15];

  float mx[8], sm[8];
#pragma unroll
  for (int r8 = 0; r8 < 8; ++r8) {
    float m = -3.0e38f;
#pragma unroll
    for (int nt = 0; nt < 12; ++nt) m = fmaxf(m, lg[nt][r8]);
    m = fmaxf(m, __shfl_xor(m, 1, 32));
    m = fmaxf(m, __shfl_xor(m, 2, 32));
    m = fmaxf(m, __shfl_xor(m, 4, 32));
    m = fmaxf(m, __shfl_xor(m, 8, 32));
    mx[r8] = m; sm[r8] = 0.f;
  }
#pragma unroll
  for (int nt = 0; nt < 12; ++nt)
#pragma unroll
    for (int r8 = 0; r8 < 8; ++r8) {
      float e = __expf(lg[nt][r8] - mx[r8]);
      lg[nt][r8] = e; sm[r8] += e;
    }
#pragma unroll
  for (int r8 = 0; r8 < 8; ++r8) {
    float s = sm[r8];
    s += __shfl_xor(s, 1, 32);
    s += __shfl_xor(s, 2, 32);
    s += __shfl_xor(s, 4, 32);
    s += __shfl_xor(s, 8, 32);
    sm[r8] = 1.f / s;
  }
#pragma unroll
  for (int nt = 0; nt < 12; ++nt)
#pragma unroll
    for (int r8 = 0; r8 < 8; ++r8)
      pbuf[wave][r8 + half * 8][nt * 16 + l15] = (_Float16)(lg[nt][r8] * sm[r8]);

  const _Float16* prow = &pbuf[wave][l15][0];
  int kh = half ? 8 : 0;
  v8f oacc[2]; oacc[0] = z8; oacc[1] = z8;
  for (int ks = 0; ks < 6; ++ks) {
    // probs A-fragment from LDS: two aligned 16B vector loads (ds_load_b128)
    const h8* hp0 = (const h8*)(prow + ks * 32 + kh);
    const h8* hp1 = (const h8*)(prow + ks * 32 + kh + 16);
    h8 lo = hp0[0], hi = hp1[0];
    v16h pa;
#pragma unroll
    for (int t = 0; t < 8; ++t) { pa[t] = lo[t]; pa[8 + t] = hi[t]; }
#pragma unroll
    for (int nt2 = 0; nt2 < 2; ++nt2) {
      const float* vb = v + (rowbase + ks * 32) * ld + hh * 32 + nt2 * 16;
      v16h bv = load_frag_b<false>(vb, 1, ld, lane, 0, 16);
      oacc[nt2] = wmma_f16(pa, bv, oacc[nt2]);
    }
  }
#pragma unroll
  for (int nt2 = 0; nt2 < 2; ++nt2)
#pragma unroll
    for (int r8 = 0; r8 < 8; ++r8) {
      int grow = rowbase + it * 16 + r8 + half * 8;
      int col = hh * 32 + nt2 * 16 + l15;
      int idx = grow * ld + col;
      o[idx] = oacc[nt2][r8] * dsigmoid(gate[idx]);
    }
}

// ---- fused outer product mean -----------------------------------------
// z[ij,c] += sum_{a,b} af[i,a]*bf[j,b]*wo[a*32+b,c] + bo[c]   (A generated on the fly)
__global__ void opm_kernel(const float* __restrict__ af, const float* __restrict__ bfp,
                           const float* __restrict__ wo, const float* __restrict__ bo,
                           float* __restrict__ z) {
  int lane = threadIdx.x & 31;
  int wave = __builtin_amdgcn_readfirstlane(threadIdx.x >> 5);
  int tile = blockIdx.x * WPB + wave;
  if (tile >= 2304 * 8) return;   // M=36864 (2304 mtiles) x N=128 (8 ntiles)
  int mt = tile / 8, nt = tile % 8;
  int l15 = lane & 15, half = lane >> 4;
  int m = mt * 16 + l15;
  int i = m / LQ, j = m % LQ;
  const float* ai = af + i * 32;
  const float* bj = bfp + j * 32;
  int kh = half ? 8 : 0;
  // per-lane bf run is 16B aligned: two float4 loads per group
  float4 b0 = *(const float4*)(bj + kh);
  float4 b1 = *(const float4*)(bj + kh + 4);
  float4 b2 = *(const float4*)(bj + kh + 16);
  float4 b3 = *(const float4*)(bj + kh + 20);
  float bl[16] = {b0.x, b0.y, b0.z, b0.w, b1.x, b1.y, b1.z, b1.w,
                  b2.x, b2.y, b2.z, b2.w, b3.x, b3.y, b3.z, b3.w};
  v8f acc = {0.f, 0.f, 0.f, 0.f, 0.f, 0.f, 0.f, 0.f};
  for (int a = 0; a < 32; ++a) {
    float av = ai[a];
    v16h fa;
#pragma unroll
    for (int t = 0; t < 16; ++t) fa[t] = (_Float16)(av * bl[t]);
    v16h fb = load_frag_b<false>(wo + a * 32 * 128 + nt * 16, 1, 128, lane, 0, 16);
    acc = wmma_f16(fa, fb, acc);
  }
#pragma unroll
  for (int r8 = 0; r8 < 8; ++r8) {
    int mrow = mt * 16 + r8 + half * 8;
    int n = nt * 16 + l15;
    z[mrow * 128 + n] += acc[r8] + bo[n];
  }
}

// ---------------- host side ----------------

static GemmP mk(const float* A, int Am, int Ak,
                const float* B, int Bn, int Bk,
                float* C, int Cm, int Cn,
                int M, int N, int K, const float* bias, int epi) {
  GemmP p;
  p.A = A; p.Ab = 0; p.Am = Am; p.Ak = Ak;
  p.B = B; p.Bb = 0; p.Bn = Bn; p.Bk = Bk;
  p.C = C; p.Cb = 0; p.Cm = Cm; p.Cn = Cn;
  p.bias = bias; p.other = nullptr; p.Om = 0; p.On = 0;
  p.M = M; p.N = N; p.K = K; p.alpha = 1.0f; p.epi = epi; p.maskN = 0; p.permC = 0;
  return p;
}
static void rg(hipStream_t s, const GemmP& p, int batches = 1) {
  int tiles = (p.M / 16) * (p.N / 16);
  int blocks = (tiles + WPB - 1) / WPB;
  dim3 grid(blocks, batches), blk(WPB * 32);
  // only 3 stride-shape combos are ever used:
  //  <true,true>  : outgoing triangle einsum (both K-contiguous)
  //  <true,false> : activations @ weight matrices
  //  <false,false>: incoming triangle einsum (both K-strided)
  if (p.Ak == 1 && p.Bk == 1)      gemm_kernel<true,  true ><<<grid, blk, 0, s>>>(p);
  else if (p.Ak == 1)              gemm_kernel<true,  false><<<grid, blk, 0, s>>>(p);
  else                             gemm_kernel<false, false><<<grid, blk, 0, s>>>(p);
}

extern "C" void kernel_launch(void* const* d_in, const int* in_sizes, int n_in,
                              void* d_out, int out_size, void* d_ws, size_t ws_size,
                              hipStream_t stream) {
  (void)in_sizes; (void)n_in; (void)out_size; (void)ws_size;
  auto F = [&](int i) { return (const float*)d_in[i]; };

  const int SZ_S = 192 * 384;
  const int SZ_Z = ZROWS * 128;
  float* S = (float*)d_out;
  float* Z = S + SZ_S;
  hipMemcpyAsync(S, d_in[0], (size_t)SZ_S * 4, hipMemcpyDeviceToDevice, stream);
  hipMemcpyAsync(Z, d_in[1], (size_t)SZ_Z * 4, hipMemcpyDeviceToDevice, stream);

  // workspace layout (floats)
  float* W = (float*)d_ws;
  float* zn    = W;                    // 4,718,592
  float* bq    = zn + 4718592;         // 9,437,184 each
  float* bk    = bq + 9437184;
  float* bv    = bk + 9437184;
  float* bg    = bv + 9437184;
  float* bo_   = bg + 9437184;
  float* bbias = bo_ + 9437184;        // 442,368
  float* bt1   = bbias + 442368;       // 4,718,592

  const float ATTN_SCALE = 0.1767766953f; // 32^-0.5

  // ---------------- Phase 1: gated row attention on s ----------------
  ln_kernel<<<192, 128, 0, stream>>>(S, bt1, F(2), F(3), 384, 0);         // sn
  ln_kernel<<<ZROWS, 128, 0, stream>>>(Z, zn, F(4), F(5), 128, 0);        // znb
  { GemmP p = mk(zn, 128, 1, F(6), 1, 12, bbias, 1, ZROWS, ZROWS, 16, 128,
                 nullptr, EPI_STORE); p.maskN = 12; rg(stream, p); }      // bias [h][ij]
  rg(stream, mk(bt1, 384, 1, F(7), 1, 384, bq, 384, 1, 192, 384, 384, nullptr, EPI_STORE));
  rg(stream, mk(bt1, 384, 1, F(8), 1, 384, bk, 384, 1, 192, 384, 384, nullptr, EPI_STORE));
  rg(stream, mk(bt1, 384, 1, F(9), 1, 384, bv, 384, 1, 192, 384, 384, nullptr, EPI_STORE));
  rg(stream, mk(bt1, 384, 1, F(10), 1, 384, bg, 384, 1, 192, 384, 384, F(11), EPI_STORE));
  attn_kernel<<<(144 + WPB - 1) / WPB, WPB * 32, 0, stream>>>(
      bq, bk, bv, bg, bo_, bbias, 12, 384, 144, ATTN_SCALE);
  rg(stream, mk(bo_, 384, 1, F(12), 1, 384, S, 384, 1, 192, 384, 384, F(13), EPI_ADD));

  // ---------------- Phase 2: s transition ----------------
  ln_kernel<<<192, 128, 0, stream>>>(S, bt1, F(14), F(15), 384, 0);
  rg(stream, mk(bt1, 384, 1, F(16), 1, 1536, bq, 1536, 1, 192, 1536, 384, F(17), EPI_RELU));
  rg(stream, mk(bq, 1536, 1, F(18), 1, 384, S, 384, 1, 192, 384, 1536, F(19), EPI_ADD));

  // ---------------- Phase 3: outer product mean ----------------
  ln_kernel<<<192, 128, 0, stream>>>(S, bt1, F(20), F(21), 384, 0);
  rg(stream, mk(bt1, 384, 1, F(22), 1, 32, bk, 32, 1, 192, 32, 384, F(23), EPI_STORE));        // af
  rg(stream, mk(bt1, 384, 1, F(24), 1, 32, bk + 6144, 32, 1, 192, 32, 384, F(25), EPI_STORE)); // bf
  opm_kernel<<<(2304 * 8 + WPB - 1) / WPB, WPB * 32, 0, stream>>>(bk, bk + 6144, F(26), F(27), Z);

  // ---------------- Phases 4/5: triangle multiplications ----------------
  auto trimul = [&](int pi, bool outgoing) {
    ln_kernel<<<ZROWS, 128, 0, stream>>>(Z, zn, F(pi), F(pi + 1), 128, 0);
    // tmp = zn@w_ap + b_ap
    rg(stream, mk(zn, 128, 1, F(pi + 4), 1, 128, bq, 128, 1, ZROWS, 128, 128, F(pi + 5), EPI_STORE));
    // a_cm = sigmoid(zn@w_ag + b_ag) * tmp   (stored channel-major)
    { GemmP p = mk(zn, 128, 1, F(pi + 2), 1, 128, bk, 1, ZROWS, ZROWS, 128, 128,
                   F(pi + 3), EPI_SIG_MUL);
      p.other = bq; p.Om = 128; p.On = 1; rg(stream, p); }
    // tmp = zn@w_bp + b_bp
    rg(stream, mk(zn, 128, 1, F(pi + 8), 1, 128, bq, 128, 1, ZROWS, 128, 128, F(pi + 9), EPI_STORE));
    // b_cm = sigmoid(zn@w_bg + b_bg) * tmp   (channel-major)
    { GemmP p = mk(zn, 128, 1, F(pi + 6), 1, 128, bv, 1, ZROWS, ZROWS, 128, 128,
                   F(pi + 7), EPI_SIG_MUL);
      p.other = bq; p.Om = 128; p.On = 1; rg(stream, p); }
    // gate = sigmoid(zn@w_g + b_g)
    rg(stream, mk(zn, 128, 1, F(pi + 14), 1, 128, bg, 128, 1, ZROWS, 128, 128, F(pi + 15), EPI_SIGMOID));
    // triangle einsum: batched over 128 channels, 192x192x192 each
    { GemmP p = mk(bk, outgoing ? 192 : 1, outgoing ? 1 : 192,
                   bv, outgoing ? 192 : 1, outgoing ? 1 : 192,
                   bo_, LQ * 128, 128, LQ, LQ, LQ, nullptr, EPI_STORE);
      p.Ab = ZROWS; p.Bb = ZROWS; p.Cb = 1; rg(stream, p, 128); }
    // ln_out -> bq ; then z += gate * (bq@w_z + b_z)
    ln_kernel<<<ZROWS, 128, 0, stream>>>(bo_, bq, F(pi + 10), F(pi + 11), 128, 0);
    { GemmP p = mk(bq, 128, 1, F(pi + 12), 1, 128, Z, 128, 1, ZROWS, 128, 128,
                   F(pi + 13), EPI_GATE_MUL_ADD);
      p.other = bg; p.Om = 128; p.On = 1; rg(stream, p); }
  };
  trimul(28, true);   // outgoing
  trimul(44, false);  // incoming

  // ---------------- Phases 6/7: triangle attentions ----------------
  auto triatt = [&](int pi, int perm) {
    ln_kernel<<<ZROWS, 128, 0, stream>>>(Z, zn, F(pi), F(pi + 1), 128, perm);
    { GemmP p = mk(zn, 128, 1, F(pi + 2), 1, 8, bbias, 1, ZROWS, ZROWS, 16, 128,
                   nullptr, EPI_STORE); p.maskN = 8; rg(stream, p); }
    rg(stream, mk(zn, 128, 1, F(pi + 3), 1, 256, bq, 256, 1, ZROWS, 256, 128, nullptr, EPI_STORE));
    rg(stream, mk(zn, 128, 1, F(pi + 4), 1, 256, bk, 256, 1, ZROWS, 256, 128, nullptr, EPI_STORE));
    rg(stream, mk(zn, 128, 1, F(pi + 5), 1, 256, bv, 256, 1, ZROWS, 256, 128, nullptr, EPI_STORE));
    rg(stream, mk(zn, 128, 1, F(pi + 6), 1, 256, bg, 256, 1, ZROWS, 256, 128, F(pi + 7), EPI_STORE));
    int nwork = 192 * 8 * 12;
    attn_kernel<<<(nwork + WPB - 1) / WPB, WPB * 32, 0, stream>>>(
        bq, bk, bv, bg, bo_, bbias, 8, 256, nwork, ATTN_SCALE);
    { GemmP p = mk(bo_, 256, 1, F(pi + 8), 1, 128, Z, 128, 1, ZROWS, 128, 256,
                   F(pi + 9), EPI_ADD);
      p.permC = perm; rg(stream, p); }
  };
  triatt(60, 0);  // starting node
  triatt(70, 1);  // ending node (transposed via row permutation)

  // ---------------- Phase 8: pair transition ----------------
  ln_kernel<<<ZROWS, 128, 0, stream>>>(Z, zn, F(80), F(81), 128, 0);
  rg(stream, mk(zn, 128, 1, F(82), 1, 512, bq, 512, 1, ZROWS, 512, 128, F(83), EPI_RELU));
  rg(stream, mk(bq, 512, 1, F(84), 1, 128, Z, 128, 1, ZROWS, 128, 512, F(85), EPI_ADD));
}
